// Transformer_18571438588025
// MI455X (gfx1250) — matrix operand order repaired and checked by reading, not compiled, not run on hardware
//
#include <hip/hip_runtime.h>

typedef unsigned short u16;
typedef __attribute__((ext_vector_type(16))) __bf16 v16bf;
typedef __attribute__((ext_vector_type(8)))  float  v8f;

#define S_LEN  4096
#define DIM    1024
#define DEPTH  8
#define NHEAD  16
#define DHEAD  64
#define HIDDEN 4096
#define VOCAB  8192

__device__ __forceinline__ u16 f2bf(float f) {
  unsigned int u = __builtin_bit_cast(unsigned int, f);
  u += 0x7FFFu + ((u >> 16) & 1u);            // round-to-nearest-even
  return (u16)(u >> 16);
}
__device__ __forceinline__ float bf2f(u16 h) {
  unsigned int u = ((unsigned int)h) << 16;
  return __builtin_bit_cast(float, u);
}

union FragAB { uint4 u[2]; v16bf v; };

__device__ __forceinline__ v8f zero8() {
  v8f z;
#pragma unroll
  for (int i = 0; i < 8; ++i) z[i] = 0.f;
  return z;
}

__device__ __forceinline__ v8f wmma_bf16(const FragAB& a, const FragAB& b, v8f c) {
  return __builtin_amdgcn_wmma_f32_16x16x32_bf16(false, a.v, false, b.v,
                                                 (short)0, c, false, false);
}

// CDNA5 async copy: global -> LDS without touching VGPRs (ASYNCcnt-tracked).
__device__ __forceinline__ void async_b128(unsigned lds_off, const u16* gptr) {
  asm volatile("global_load_async_to_lds_b128 %0, %1, off"
               :: "v"(lds_off), "v"(gptr) : "memory");
}
__device__ __forceinline__ void wait_async_le4() {
  asm volatile("s_wait_asynccnt 0x4" ::: "memory");
}
__device__ __forceinline__ void wait_async_0() {
  asm volatile("s_wait_asynccnt 0x0" ::: "memory");
}

// ---------------------------------------------------------------- fp32 -> bf16
__global__ void __launch_bounds__(256) k_cvt(const float* __restrict__ in,
                                             u16* __restrict__ out, int n) {
  int i = (blockIdx.x * 256 + threadIdx.x) * 4;
  if (i >= n) return;
  float4 f = *(const float4*)(in + i);
  union { u16 h[4]; uint2 u; } pk;
  pk.h[0] = f2bf(f.x); pk.h[1] = f2bf(f.y);
  pk.h[2] = f2bf(f.z); pk.h[3] = f2bf(f.w);
  *(uint2*)(out + i) = pk.u;
}

// ---------------------------------------------------------------- embedding
__global__ void __launch_bounds__(256) k_embed(const int* __restrict__ seq,
                                               const int* __restrict__ act,
                                               const float* __restrict__ emb,
                                               const float* __restrict__ actw,
                                               float* __restrict__ X) {
  int s  = blockIdx.x;
  int d0 = threadIdx.x * 4;
  int tok = seq[s];
  int a   = act[s >> 6];                 // BLOCK = 64
  const float* ep = emb  + (size_t)tok * DIM + d0;
  const float* ap = actw + (size_t)a   * DIM + d0;
  float*       xp = X    + (size_t)s   * DIM + d0;
#pragma unroll
  for (int j = 0; j < 4; ++j) {
    int d = d0 + j;
    float freq = __expf(-9.210340371976184f * (float)(d & ~1) * (1.f / DIM));
    float ang  = (float)s * freq;
    float pe   = (d & 1) ? __cosf(ang) : __sinf(ang);
    xp[j] = ep[j] + pe + ap[j];
  }
}

// ---------------------------------------------------------------- RMSNorm (wave per row)
__global__ void __launch_bounds__(256) k_rmsnorm(const float* __restrict__ X,
                                                 const float* __restrict__ w,
                                                 const float* __restrict__ b,
                                                 u16* __restrict__ Y) {
  int lane = threadIdx.x & 31;
  int wv   = threadIdx.x >> 5;
  int row  = blockIdx.x * 8 + wv;
  const float* xp = X + (size_t)row * DIM;
  float xv[32];
  float ss = 0.f;
#pragma unroll
  for (int i = 0; i < 32; ++i) {
    xv[i] = xp[lane + 32 * i];
    ss += xv[i] * xv[i];
  }
#pragma unroll
  for (int off = 16; off > 0; off >>= 1) ss += __shfl_xor(ss, off, 32);
  float inv = rsqrtf(ss * (1.f / DIM) + 1e-5f);
  u16* yp = Y + (size_t)row * DIM;
#pragma unroll
  for (int i = 0; i < 32; ++i) {
    int d = lane + 32 * i;
    yp[d] = f2bf(xv[i] * inv * w[d] + b[d]);
  }
}

// ---------------------------------------------------------------- GEMM: out = A[M,K] * W[N,K]^T (+bias)(+resid)
// Double-buffered LDS staging fed by global_load_async_to_lds_b128.
__global__ void __launch_bounds__(256) k_gemm(const u16* __restrict__ A,
                                              const u16* __restrict__ W,
                                              const float* __restrict__ bias,
                                              const float* __restrict__ resid,
                                              float* __restrict__ outF,
                                              u16* __restrict__ outB,
                                              int M, int N, int K) {
  __shared__ alignas(16) u16 As[2][128 * 40];   // 128 rows x 32 k, stride 40
  __shared__ alignas(16) u16 Bs[2][128 * 40];
  const int tid  = threadIdx.x;
  const int lane = tid & 31;
  const int wv   = tid >> 5;                 // 8 waves
  const int wm   = wv & 3;                   // 4 M strips of 32
  const int wn   = wv >> 2;                  // 2 N strips of 64
  const int l16  = lane & 15;
  const int ah   = (lane < 16) ? 0 : 8;      // A-frag K half
  const int bh   = (lane < 16) ? 0 : 16;     // B-frag K half
  const int m0   = blockIdx.y * 128;
  const int n0   = blockIdx.x * 128;

  // this thread's two 16B staging chunks: rows r, r+64 at column cc
  const int r  = tid >> 2;
  const int cc = (tid & 3) * 8;
  const u16* Ap0 = A + (size_t)(m0 + r) * K + cc;
  const u16* Ap1 = A + (size_t)(m0 + r + 64) * K + cc;
  const u16* Bp0 = W + (size_t)(n0 + r) * K + cc;
  const u16* Bp1 = W + (size_t)(n0 + r + 64) * K + cc;
  const int lo0 = r * 40 + cc;
  const int lo1 = (r + 64) * 40 + cc;

  auto prefetch = [&](int kt, int buf) {
    async_b128((unsigned)(size_t)&As[buf][lo0], Ap0 + kt);
    async_b128((unsigned)(size_t)&As[buf][lo1], Ap1 + kt);
    async_b128((unsigned)(size_t)&Bs[buf][lo0], Bp0 + kt);
    async_b128((unsigned)(size_t)&Bs[buf][lo1], Bp1 + kt);
  };

  v8f acc[2][4];
#pragma unroll
  for (int mf = 0; mf < 2; ++mf)
#pragma unroll
    for (int nf = 0; nf < 4; ++nf) acc[mf][nf] = zero8();

  const int nk = K >> 5;
  prefetch(0, 0);
  for (int it = 0; it < nk; ++it) {
    const int buf = it & 1;
    if (it + 1 < nk) {
      prefetch((it + 1) << 5, buf ^ 1);
      wait_async_le4();                      // stage `it` (first 4 ops) landed
    } else {
      wait_async_0();
    }
    __syncthreads();
    FragAB af[2], bf[4];
#pragma unroll
    for (int mf = 0; mf < 2; ++mf) {
      int base = (wm * 32 + mf * 16 + l16) * 40;
      af[mf].u[0] = *(const uint4*)&As[buf][base + ah];
      af[mf].u[1] = *(const uint4*)&As[buf][base + 16 + ah];
    }
#pragma unroll
    for (int nf = 0; nf < 4; ++nf) {
      int base = (wn * 64 + nf * 16 + l16) * 40;
      bf[nf].u[0] = *(const uint4*)&Bs[buf][base + ah];
      bf[nf].u[1] = *(const uint4*)&Bs[buf][base + 16 + ah];
    }
#pragma unroll
    for (int mf = 0; mf < 2; ++mf)
#pragma unroll
      for (int nf = 0; nf < 4; ++nf)
        acc[mf][nf] = wmma_bf16(af[mf], bf[nf], acc[mf][nf]);
    __syncthreads();                         // all reads done before buf reuse
  }

#pragma unroll
  for (int mf = 0; mf < 2; ++mf)
#pragma unroll
    for (int nf = 0; nf < 4; ++nf) {
      int col = n0 + wn * 64 + nf * 16 + l16;
      float bv = bias ? bias[col] : 0.f;
#pragma unroll
      for (int i = 0; i < 8; ++i) {
        int row = m0 + wm * 32 + mf * 16 + ((lane < 16) ? i : 8 + i);
        float v = acc[mf][nf][i] + bv;
        if (resid) v += resid[(size_t)row * N + col];
        if (outF) outF[(size_t)row * N + col] = v;
        if (outB) outB[(size_t)row * N + col] = f2bf(v);
      }
    }
}

// ---------------------------------------------------------------- bf16 transpose (V -> V^T)
__global__ void __launch_bounds__(256) k_transpose(const u16* __restrict__ in,
                                                   u16* __restrict__ out,
                                                   int rows, int cols) {
  __shared__ u16 t[32][33];
  int c0 = blockIdx.x * 32, r0 = blockIdx.y * 32;
  int tx = threadIdx.x, ty = threadIdx.y;
#pragma unroll
  for (int j = 0; j < 4; ++j)
    t[ty + 8 * j][tx] = in[(size_t)(r0 + ty + 8 * j) * cols + c0 + tx];
  __syncthreads();
#pragma unroll
  for (int j = 0; j < 4; ++j)
    out[(size_t)(c0 + ty + 8 * j) * rows + r0 + tx] = t[tx][ty + 8 * j];
}

// ---------------------------------------------------------------- flash attention, block-causal (BLOCK==64)
__global__ void __launch_bounds__(128) k_attn(const u16* __restrict__ Q,
                                              const u16* __restrict__ Km,
                                              const u16* __restrict__ VT,
                                              u16* __restrict__ O) {
  __shared__ alignas(16) u16 P[4][16][72];   // per-wave 16x64 P strip (padded)
  const int tid  = threadIdx.x;
  const int lane = tid & 31;
  const int w    = tid >> 5;                 // 4 waves, 16 q-rows each
  const int l16  = lane & 15;
  const int ah   = (lane < 16) ? 0 : 8;
  const int bh   = (lane < 16) ? 0 : 8;
  const int h    = blockIdx.y;
  const int qb   = blockIdx.x;
  const int mb   = qb * 64 + w * 16;
  const int hd   = h * DHEAD;

  FragAB qa[2];
  {
    const u16* qp = Q + (size_t)(mb + l16) * DIM + hd;
    qa[0].u[0] = *(const uint4*)(qp + ah);
    qa[0].u[1] = *(const uint4*)(qp + 16 + ah);
    qa[1].u[0] = *(const uint4*)(qp + 32 + ah);
    qa[1].u[1] = *(const uint4*)(qp + 48 + ah);
  }

  float mrow[8], lrow[8];
  v8f of[4];
#pragma unroll
  for (int i = 0; i < 8; ++i) { mrow[i] = -3.0e38f; lrow[i] = 0.f; }
#pragma unroll
  for (int nf = 0; nf < 4; ++nf) of[nf] = zero8();

  for (int kb = 0; kb <= qb; ++kb) {
    // S = Q K^T * scale
    v8f sf[4];
#pragma unroll
    for (int nf = 0; nf < 4; ++nf) {
      const u16* kp = Km + (size_t)(kb * 64 + nf * 16 + l16) * DIM + hd;
      FragAB b0, b1;
      b0.u[0] = *(const uint4*)(kp + bh);
      b0.u[1] = *(const uint4*)(kp + bh + 16);
      b1.u[0] = *(const uint4*)(kp + 32 + bh);
      b1.u[1] = *(const uint4*)(kp + 32 + bh + 16);
      v8f z = zero8();
      z = wmma_bf16(qa[0], b0, z);
      z = wmma_bf16(qa[1], b1, z);
#pragma unroll
      for (int i = 0; i < 8; ++i) z[i] *= 0.125f;   // 1/sqrt(64)
      sf[nf] = z;
    }
    // online softmax (rows split over 16-lane groups)
    float mx[8];
#pragma unroll
    for (int i = 0; i < 8; ++i)
      mx[i] = fmaxf(fmaxf(sf[0][i], sf[1][i]), fmaxf(sf[2][i], sf[3][i]));
#pragma unroll
    for (int off = 1; off < 16; off <<= 1)
#pragma unroll
      for (int i = 0; i < 8; ++i) mx[i] = fmaxf(mx[i], __shfl_xor(mx[i], off, 32));
    float corr[8], rs[8];
#pragma unroll
    for (int i = 0; i < 8; ++i) {
      float mn = fmaxf(mrow[i], mx[i]);
      corr[i] = __expf(mrow[i] - mn);
      mrow[i] = mn;
      rs[i] = 0.f;
    }
#pragma unroll
    for (int nf = 0; nf < 4; ++nf)
#pragma unroll
      for (int i = 0; i < 8; ++i) {
        float p = __expf(sf[nf][i] - mrow[i]);
        rs[i] += p;
        P[w][(lane < 16) ? i : 8 + i][nf * 16 + l16] = f2bf(p);
      }
#pragma unroll
    for (int off = 1; off < 16; off <<= 1)
#pragma unroll
      for (int i = 0; i < 8; ++i) rs[i] += __shfl_xor(rs[i], off, 32);
#pragma unroll
    for (int i = 0; i < 8; ++i) lrow[i] = lrow[i] * corr[i] + rs[i];
#pragma unroll
    for (int nf = 0; nf < 4; ++nf)
#pragma unroll
      for (int i = 0; i < 8; ++i) of[nf][i] *= corr[i];
    __syncthreads();
    // O += P V  (P re-read from LDS in A layout, V from V^T contiguously)
    FragAB pa[2];
    pa[0].u[0] = *(const uint4*)&P[w][l16][ah];
    pa[0].u[1] = *(const uint4*)&P[w][l16][16 + ah];
    pa[1].u[0] = *(const uint4*)&P[w][l16][32 + ah];
    pa[1].u[1] = *(const uint4*)&P[w][l16][48 + ah];
#pragma unroll
    for (int nf = 0; nf < 4; ++nf) {
      const u16* vp = VT + (size_t)(hd + nf * 16 + l16) * S_LEN + kb * 64;
      FragAB v0, v1;
      v0.u[0] = *(const uint4*)(vp + bh);
      v0.u[1] = *(const uint4*)(vp + bh + 16);
      v1.u[0] = *(const uint4*)(vp + 32 + bh);
      v1.u[1] = *(const uint4*)(vp + 32 + bh + 16);
      of[nf] = wmma_bf16(pa[0], v0, of[nf]);
      of[nf] = wmma_bf16(pa[1], v1, of[nf]);
    }
    __syncthreads();
  }
#pragma unroll
  for (int nf = 0; nf < 4; ++nf)
#pragma unroll
    for (int i = 0; i < 8; ++i) {
      int row = mb + ((lane < 16) ? i : 8 + i);
      O[(size_t)row * DIM + hd + nf * 16 + l16] = f2bf(of[nf][i] / lrow[i]);
    }
}

// ---------------------------------------------------------------- GeGLU (tanh gelu)
__global__ void __launch_bounds__(256) k_geglu(const u16* __restrict__ F,
                                               u16* __restrict__ A) {
  int idx = blockIdx.x * 256 + threadIdx.x;    // S*HIDDEN
  int s = idx >> 12, j = idx & (HIDDEN - 1);
  float h = bf2f(F[(size_t)s * (2 * HIDDEN) + j]);
  float g = bf2f(F[(size_t)s * (2 * HIDDEN) + HIDDEN + j]);
  float t = tanhf(0.7978845608028654f * (h + 0.044715f * h * h * h));
  A[idx] = f2bf(0.5f * h * (1.f + t) * g);
}

// ---------------------------------------------------------------- host orchestration
static inline void cvt(const float* in, u16* out, int n, hipStream_t s) {
  k_cvt<<<(n / 4 + 255) / 256, 256, 0, s>>>(in, out, n);
}

extern "C" void kernel_launch(void* const* d_in, const int* in_sizes, int n_in,
                              void* d_out, int out_size, void* d_ws, size_t ws_size,
                              hipStream_t stream) {
  const int*   seq    = (const int*)d_in[0];
  const int*   action = (const int*)d_in[1];
  const float* emb_w  = (const float*)d_in[2];
  const float* act_w  = (const float*)d_in[3];
  const float* ln1_w  = (const float*)d_in[4];
  const float* ln1_b  = (const float*)d_in[5];
  const float* wq     = (const float*)d_in[6];
  const float* wk     = (const float*)d_in[7];
  const float* wvp    = (const float*)d_in[8];
  const float* wo     = (const float*)d_in[9];
  const float* ln2_w  = (const float*)d_in[10];
  const float* ln2_b  = (const float*)d_in[11];
  const float* fc1_w  = (const float*)d_in[12];
  const float* fc1_b  = (const float*)d_in[13];
  const float* fc2_w  = (const float*)d_in[14];
  const float* fc2_b  = (const float*)d_in[15];
  const float* lnf_w  = (const float*)d_in[16];
  const float* lnf_b  = (const float*)d_in[17];
  const float* head_w = (const float*)d_in[18];
  const float* head_b = (const float*)d_in[19];

  char* ws = (char*)d_ws;
  size_t off = 0;
  auto alloc = [&](size_t bytes) {
    void* p = ws + off;
    off = (off + bytes + 255) & ~(size_t)255;
    return p;
  };
  u16* wqB   = (u16*)alloc((size_t)DIM * DIM * 2);
  u16* wkB   = (u16*)alloc((size_t)DIM * DIM * 2);
  u16* wvB   = (u16*)alloc((size_t)DIM * DIM * 2);
  u16* woB   = (u16*)alloc((size_t)DIM * DIM * 2);
  u16* fc1B  = (u16*)alloc((size_t)2 * HIDDEN * DIM * 2);
  u16* fc2B  = (u16*)alloc((size_t)DIM * HIDDEN * 2);
  u16* headB = (u16*)alloc((size_t)VOCAB * DIM * 2);
  float* X   = (float*)alloc((size_t)S_LEN * DIM * 4);
  u16* YB    = (u16*)alloc((size_t)S_LEN * DIM * 2);
  u16* QB    = (u16*)alloc((size_t)S_LEN * DIM * 2);
  u16* KB    = (u16*)alloc((size_t)S_LEN * DIM * 2);
  u16* VB    = (u16*)alloc((size_t)S_LEN * DIM * 2);
  u16* VTB   = (u16*)alloc((size_t)DIM * S_LEN * 2);
  u16* ATTB  = (u16*)alloc((size_t)S_LEN * DIM * 2);
  u16* FF1B  = (u16*)alloc((size_t)S_LEN * 2 * HIDDEN * 2);
  u16* ACTB  = (u16*)alloc((size_t)S_LEN * HIDDEN * 2);
  (void)ws_size; (void)in_sizes; (void)n_in; (void)out_size;

  auto gemm = [&](const u16* A, const u16* W, const float* bias, const float* res,
                  float* oF, u16* oB, int M, int N, int K) {
    k_gemm<<<dim3(N / 128, M / 128), 256, 0, stream>>>(A, W, bias, res, oF, oB, M, N, K);
  };

  cvt(head_w, headB, VOCAB * DIM, stream);
  k_embed<<<S_LEN, 256, 0, stream>>>(seq, action, emb_w, act_w, X);

  for (int l = 0; l < DEPTH; ++l) {
    cvt(wq + (size_t)l * DIM * DIM, wqB, DIM * DIM, stream);
    cvt(wk + (size_t)l * DIM * DIM, wkB, DIM * DIM, stream);
    cvt(wvp + (size_t)l * DIM * DIM, wvB, DIM * DIM, stream);
    cvt(wo + (size_t)l * DIM * DIM, woB, DIM * DIM, stream);
    cvt(fc1_w + (size_t)l * 2 * HIDDEN * DIM, fc1B, 2 * HIDDEN * DIM, stream);
    cvt(fc2_w + (size_t)l * DIM * HIDDEN, fc2B, DIM * HIDDEN, stream);

    k_rmsnorm<<<S_LEN / 8, 256, 0, stream>>>(X, ln1_w + l * DIM, ln1_b + l * DIM, YB);
    gemm(YB, wqB, nullptr, nullptr, nullptr, QB, S_LEN, DIM, DIM);
    gemm(YB, wkB, nullptr, nullptr, nullptr, KB, S_LEN, DIM, DIM);
    gemm(YB, wvB, nullptr, nullptr, nullptr, VB, S_LEN, DIM, DIM);
    k_transpose<<<dim3(DIM / 32, S_LEN / 32), dim3(32, 8), 0, stream>>>(VB, VTB, S_LEN, DIM);
    k_attn<<<dim3(S_LEN / 64, NHEAD), 128, 0, stream>>>(QB, KB, VTB, ATTB);
    gemm(ATTB, woB, nullptr, X, X, nullptr, S_LEN, DIM, DIM);   // x += o @ Wo^T

    k_rmsnorm<<<S_LEN / 8, 256, 0, stream>>>(X, ln2_w + l * DIM, ln2_b + l * DIM, YB);
    gemm(YB, fc1B, fc1_b + (size_t)l * 2 * HIDDEN, nullptr, nullptr, FF1B,
         S_LEN, 2 * HIDDEN, DIM);
    k_geglu<<<(S_LEN * HIDDEN) / 256, 256, 0, stream>>>(FF1B, ACTB);
    gemm(ACTB, fc2B, fc2_b + (size_t)l * DIM, X, X, nullptr, S_LEN, DIM, HIDDEN);
  }

  k_rmsnorm<<<S_LEN / 8, 256, 0, stream>>>(X, lnf_w, lnf_b, YB);
  gemm(YB, headB, head_b, nullptr, (float*)d_out, nullptr, S_LEN, VOCAB, DIM);
}
